// GCNEncoder_20779051778306
// MI455X (gfx1250) — matrix operand order repaired and checked by reading, not compiled, not run on hardware
//
#include <hip/hip_runtime.h>

typedef __attribute__((ext_vector_type(16))) __bf16 v16bf;
typedef __attribute__((ext_vector_type(8)))  __bf16 v8bf;
typedef __attribute__((ext_vector_type(8)))  float  v8f;
typedef __attribute__((ext_vector_type(4)))  int    v4i;

#define IN_DIM 2063
#define KPAD   2080   // 2063 padded to multiple of 32
#define NGRAPH 64

// CDNA5 async global->LDS copy path (tracked by ASYNCcnt), with safe fallback.
#if defined(__AMDGCN__) && __has_builtin(__builtin_amdgcn_global_load_async_to_lds_b128) && \
    __has_builtin(__builtin_amdgcn_s_wait_asynccnt)
#define USE_ASYNC_LDS 1
#else
#define USE_ASYNC_LDS 0
#endif

static __device__ __forceinline__ void atomAddF(float* p, float v) {
  __hip_atomic_fetch_add(p, v, __ATOMIC_RELAXED, __HIP_MEMORY_SCOPE_AGENT);
}

// ---------------------------------------------------------------- utilities
__global__ void fill_f32(float* __restrict__ p, float v, long long n) {
  long long i = blockIdx.x * (long long)blockDim.x + threadIdx.x;
  if (i < n) p[i] = v;
}

__global__ void cast_f32_bf16(const float* __restrict__ in, __bf16* __restrict__ out, long long n) {
  long long i = blockIdx.x * (long long)blockDim.x + threadIdx.x;
  if (i < n) out[i] = (__bf16)in[i];
}

// x [N x 2063] f32 -> xb [N x 2080] bf16, zero-padded K tail
__global__ void cast_x(const float* __restrict__ x, __bf16* __restrict__ xb, int N) {
  long long i = blockIdx.x * (long long)blockDim.x + threadIdx.x;
  long long total = (long long)N * KPAD;
  if (i >= total) return;
  int col = (int)(i % KPAD);
  long long row = i / KPAD;
  float v = (col < IN_DIM) ? x[row * (long long)IN_DIM + col] : 0.0f;
  xb[i] = (__bf16)v;
}

// W_l1 [2063x256], W_g1 [2063x256] -> fused bf16 [2080 x 512]
__global__ void cast_w1(const float* __restrict__ Wl, const float* __restrict__ Wg,
                        __bf16* __restrict__ Wb) {
  long long i = blockIdx.x * (long long)blockDim.x + threadIdx.x;
  if (i >= (long long)KPAD * 512) return;
  int c = (int)(i % 512);
  int k = (int)(i / 512);
  float v = 0.0f;
  if (k < IN_DIM) v = (c < 256) ? Wl[(long long)k * 256 + c] : Wg[(long long)k * 256 + (c - 256)];
  Wb[i] = (__bf16)v;
}

// ---------------------------------------------------------------- degrees
__global__ void deg_edges(const int* __restrict__ dst, long long E, float* __restrict__ deg) {
  long long e = blockIdx.x * (long long)blockDim.x + threadIdx.x;
  if (e < E) atomAddF(&deg[dst[e]], 1.0f);
}

__global__ void rsqrt_inplace(float* __restrict__ d, int N) {
  int i = blockIdx.x * blockDim.x + threadIdx.x;
  if (i < N) d[i] = rsqrtf(d[i]);
}

// ---------------------------------------------------------------- WMMA GEMM
// C[M x N] = A[M x K](bf16, lda) @ B[K x N](bf16, ldb), f32 accum.
// Block tile 128x128, BK=32, 8 waves (wave32), wave tile 32x64 (2x4 WMMA frags).
// A tile staged via GLOBAL_LOAD_ASYNC_TO_LDS_B128 on full M-tiles (ASYNCcnt
// tracked); B tile transposed through VGPRs into LDS.
__global__ __launch_bounds__(256)
void gemm_bf16(const __bf16* __restrict__ A, int lda,
               const __bf16* __restrict__ B, int ldb,
               float* __restrict__ C, int ldc,
               int M, int K) {
  __shared__ __align__(16) __bf16 As[128 * 40];  // [row][k], stride 40 (pad vs banks)
  __shared__ __align__(16) __bf16 Bs[128 * 40];  // transposed: [col][k], stride 40

  const int tid  = threadIdx.x;
  const int lane = tid & 31;
  const int wid  = tid >> 5;
  const int wm   = wid >> 1;   // 0..3  (row group of 32)
  const int wn   = wid & 1;    // 0..1  (col group of 64)
  const int half = lane >> 4;  // 0/1
  const int r    = lane & 15;
  const int mBase = blockIdx.y * 128;
  const int nBase = blockIdx.x * 128;
  const bool fullM = (mBase + 128 <= M);

  v8f acc[2][4];
#pragma unroll
  for (int mt = 0; mt < 2; mt++)
#pragma unroll
    for (int nt = 0; nt < 4; nt++)
#pragma unroll
      for (int v = 0; v < 8; v++) acc[mt][nt][v] = 0.0f;

  for (int k0 = 0; k0 < K; k0 += 32) {
    // ---- stage A tile: 128 rows x 32 cols bf16, each thread 2 x 16B chunks
#if USE_ASYNC_LDS
    if (fullM) {
#pragma unroll
      for (int i = 0; i < 2; i++) {
        int idx = tid + i * 256;
        int row = idx >> 2;              // 0..127
        int ch  = (idx & 3) * 8;         // k-chunk of 8 elems
        const __bf16* gp = A + (size_t)(mBase + row) * lda + k0 + ch;
        __bf16* lp = As + row * 40 + ch;
        __builtin_amdgcn_global_load_async_to_lds_b128(
            (__attribute__((address_space(1))) v4i*)(void*)gp,
            (__attribute__((address_space(3))) v4i*)(void*)lp, 0, 0);
      }
    } else
#endif
    {
#pragma unroll
      for (int i = 0; i < 2; i++) {
        int idx = tid + i * 256;
        int row = idx >> 2;              // 0..127
        int ch  = (idx & 3) * 8;         // k-chunk of 8 elems
        int gr  = mBase + row;
        v8bf val;
#pragma unroll
        for (int j = 0; j < 8; j++) val[j] = (__bf16)0.0f;
        if (gr < M) val = *(const v8bf*)(A + (size_t)gr * lda + k0 + ch);
        *(v8bf*)(As + row * 40 + ch) = val;
      }
    }
    // ---- stage B tile transposed: 32 k-rows x 128 cols
#pragma unroll
    for (int i = 0; i < 2; i++) {
      int idx = tid + i * 256;
      int kr  = idx >> 4;              // 0..31
      int col = (idx & 15) * 8;        // 0..120
      v8bf val = *(const v8bf*)(B + (size_t)(k0 + kr) * ldb + nBase + col);
#pragma unroll
      for (int j = 0; j < 8; j++) Bs[(col + j) * 40 + kr] = val[j];
    }
    // prefetch next K-tile streams into cache while this tile computes
    if (k0 + 32 < K) {
      __builtin_prefetch(A + (size_t)(mBase + (tid >> 2)) * lda + (k0 + 32) + (tid & 3) * 8, 0, 1);
      __builtin_prefetch(B + (size_t)(k0 + 32 + (tid >> 4)) * ldb + nBase + (tid & 15) * 8, 0, 1);
    }
#if USE_ASYNC_LDS
    if (fullM) __builtin_amdgcn_s_wait_asynccnt(0);
#endif
    __syncthreads();

    union BFV { v8bf h[2]; v16bf v; };
    BFV afrag[2], bfrag[4];
    // A frag (16x32): lane half 0 -> K {0..7,16..23}; half 1 -> K {8..15,24..31}
#pragma unroll
    for (int mt = 0; mt < 2; mt++) {
      const __bf16* ap = As + (wm * 32 + mt * 16 + r) * 40 + half * 8;
      afrag[mt].h[0] = *(const v8bf*)(ap);
      afrag[mt].h[1] = *(const v8bf*)(ap + 16);
    }
    // B frag (32x16): lane half 0 -> K {0..7,16..23}; half 1 -> K {8..15,24..31}
#pragma unroll
    for (int nt = 0; nt < 4; nt++) {
      const __bf16* bp = Bs + (wn * 64 + nt * 16 + r) * 40 + half * 8;
      bfrag[nt].h[0] = *(const v8bf*)(bp);
      bfrag[nt].h[1] = *(const v8bf*)(bp + 16);
    }
#pragma unroll
    for (int mt = 0; mt < 2; mt++)
#pragma unroll
      for (int nt = 0; nt < 4; nt++)
        acc[mt][nt] = __builtin_amdgcn_wmma_f32_16x16x32_bf16(
            false, afrag[mt].v, false, bfrag[nt].v, (short)0, acc[mt][nt], false, false);
    __syncthreads();
  }

  // C layout: reg v, lanes 0-15 -> M=v, N=r; lanes 16-31 -> M=8+v, N=r
#pragma unroll
  for (int mt = 0; mt < 2; mt++) {
#pragma unroll
    for (int nt = 0; nt < 4; nt++) {
#pragma unroll
      for (int v = 0; v < 8; v++) {
        int m = mBase + wm * 32 + mt * 16 + half * 8 + v;
        int n = nBase + wn * 64 + nt * 16 + r;
        if (m < M) C[(size_t)m * ldc + n] = acc[mt][nt][v];
      }
    }
  }
}

// ---------------------------------------------------------------- aggregation
// out[dst] += h[src] * dinv[src]*dinv[dst]   (F multiple of 4)
__global__ void scatter_edges(const float* __restrict__ h, int ldh,
                              float* __restrict__ out, int ldo,
                              const int* __restrict__ src, const int* __restrict__ dst,
                              long long E, int F, const float* __restrict__ dinv) {
  long long t = blockIdx.x * (long long)blockDim.x + threadIdx.x;
  int vecs = F >> 2;
  long long e = t / vecs;
  if (e >= E) return;
  int c = (int)(t % vecs) * 4;
  int s = src[e], d = dst[e];
  float nrm = dinv[s] * dinv[d];
  float4 v = *(const float4*)(h + (size_t)s * ldh + c);
  float* op = out + (size_t)d * ldo + c;
  atomAddF(op + 0, v.x * nrm);
  atomAddF(op + 1, v.y * nrm);
  atomAddF(op + 2, v.z * nrm);
  atomAddF(op + 3, v.w * nrm);
}

// layer1 finalize: add self-loop term + bias, ReLU, emit bf16 (fused 512 cols)
__global__ void finalize1(const float* __restrict__ h1, const float* __restrict__ agg,
                          const float* __restrict__ dinv_l, const float* __restrict__ dinv_g,
                          const float* __restrict__ b_l, const float* __restrict__ b_g,
                          __bf16* __restrict__ out, int N) {
  long long i = blockIdx.x * (long long)blockDim.x + threadIdx.x;
  if (i >= (long long)N * 512) return;
  int c = (int)(i % 512);
  long long n = i / 512;
  float di = (c < 256) ? dinv_l[n] : dinv_g[n];
  float bi = (c < 256) ? b_l[c] : b_g[c - 256];
  float v = agg[i] + h1[i] * di * di + bi;
  out[i] = (__bf16)fmaxf(v, 0.0f);
}

// layer2 finalize + mean-pool numerator scatter (128 cols per branch)
__global__ void finalize2_pool(const float* __restrict__ h2, const float* __restrict__ agg,
                               const float* __restrict__ dinv, const float* __restrict__ b2,
                               const int* __restrict__ batch,
                               float* __restrict__ pooled, int colOff, int N) {
  long long i = blockIdx.x * (long long)blockDim.x + threadIdx.x;
  if (i >= (long long)N * 128) return;
  int c = (int)(i % 128);
  long long n = i / 128;
  float di = dinv[n];
  float v = agg[i] + h2[i] * di * di + b2[c];
  atomAddF(&pooled[(size_t)batch[n] * 256 + colOff + c], v);
}

__global__ void count_nodes(const int* __restrict__ batch, float* __restrict__ cnt, int N) {
  int i = blockIdx.x * blockDim.x + threadIdx.x;
  if (i < N) atomAddF(&cnt[batch[i]], 1.0f);
}

// out[g, j] = relu(b[j] + sum_k pooled[g,k]/cnt[g] * Wf[k,j])   (64 x 256 x 128)
__global__ void fuse_out(const float* __restrict__ pooled, const float* __restrict__ cnt,
                         const float* __restrict__ Wf, const float* __restrict__ bf_,
                         float* __restrict__ out) {
  int g = blockIdx.x;
  int j = threadIdx.x;
  float inv = 1.0f / fmaxf(cnt[g], 1.0f);
  float s = bf_[j];
  for (int k = 0; k < 256; k++) s = fmaf(pooled[g * 256 + k] * inv, Wf[k * 128 + j], s);
  out[g * 128 + j] = fmaxf(s, 0.0f);
}

// ---------------------------------------------------------------- launcher
static inline int blks(long long n, int b) { return (int)((n + b - 1) / b); }

extern "C" void kernel_launch(void* const* d_in, const int* in_sizes, int n_in,
                              void* d_out, int out_size, void* d_ws, size_t ws_size,
                              hipStream_t stream) {
  const float* x      = (const float*)d_in[0];
  const int*   ei_l   = (const int*)d_in[1];
  const int*   ei_g   = (const int*)d_in[2];
  const int*   batch  = (const int*)d_in[3];
  const float* W_l1   = (const float*)d_in[4];
  const float* b_l1   = (const float*)d_in[5];
  const float* W_l2   = (const float*)d_in[6];
  const float* b_l2   = (const float*)d_in[7];
  const float* W_g1   = (const float*)d_in[8];
  const float* b_g1   = (const float*)d_in[9];
  const float* W_g2   = (const float*)d_in[10];
  const float* b_g2   = (const float*)d_in[11];
  const float* W_fuse = (const float*)d_in[12];
  const float* b_fuse = (const float*)d_in[13];
  float* out = (float*)d_out;

  const int N = in_sizes[0] / IN_DIM;
  const long long El = in_sizes[1] / 2;
  const long long Eg = in_sizes[2] / 2;
  const int* src_l = ei_l;          const int* dst_l = ei_l + El;
  const int* src_g = ei_g;          const int* dst_g = ei_g + Eg;

  // workspace carve-out
  char* w = (char*)d_ws;
  size_t off = 0;
  auto carve = [&](size_t bytes) -> void* {
    void* p = w + off;
    off += (bytes + 255) & ~(size_t)255;
    return p;
  };
  __bf16* xb    = (__bf16*)carve((size_t)N * KPAD * 2);
  __bf16* W1b   = (__bf16*)carve((size_t)KPAD * 512 * 2);
  __bf16* W2lb  = (__bf16*)carve((size_t)256 * 128 * 2);
  __bf16* W2gb  = (__bf16*)carve((size_t)256 * 128 * 2);
  float*  h1    = (float*)carve((size_t)N * 512 * 4);
  float*  agg1  = (float*)carve((size_t)N * 512 * 4);
  __bf16* h1b   = (__bf16*)carve((size_t)N * 512 * 2);
  float*  h2l   = (float*)carve((size_t)N * 128 * 4);
  float*  h2g   = (float*)carve((size_t)N * 128 * 4);
  float*  agg2l = (float*)carve((size_t)N * 128 * 4);
  float*  agg2g = (float*)carve((size_t)N * 128 * 4);
  float*  dinvl = (float*)carve((size_t)N * 4);
  float*  dinvg = (float*)carve((size_t)N * 4);
  float*  pooled= (float*)carve((size_t)NGRAPH * 256 * 4);
  float*  cnt   = (float*)carve((size_t)NGRAPH * 4);
  (void)ws_size; (void)n_in; (void)out_size;

  const int T = 256;
  const int mTiles = (N + 127) / 128;

  // degrees (deg starts at 1.0 == self loop), then rsqrt in place -> dinv
  fill_f32<<<blks(N, T), T, 0, stream>>>(dinvl, 1.0f, N);
  fill_f32<<<blks(N, T), T, 0, stream>>>(dinvg, 1.0f, N);
  deg_edges<<<blks(El, T), T, 0, stream>>>(dst_l, El, dinvl);
  deg_edges<<<blks(Eg, T), T, 0, stream>>>(dst_g, Eg, dinvg);
  rsqrt_inplace<<<blks(N, T), T, 0, stream>>>(dinvl, N);
  rsqrt_inplace<<<blks(N, T), T, 0, stream>>>(dinvg, N);

  // bf16 casts (x padded once, weights fused)
  cast_x<<<blks((long long)N * KPAD, T), T, 0, stream>>>(x, xb, N);
  cast_w1<<<blks((long long)KPAD * 512, T), T, 0, stream>>>(W_l1, W_g1, W1b);
  cast_f32_bf16<<<blks(256 * 128, T), T, 0, stream>>>(W_l2, W2lb, 256 * 128);
  cast_f32_bf16<<<blks(256 * 128, T), T, 0, stream>>>(W_g2, W2gb, 256 * 128);

  // fused layer-1 GEMM: h1[N x 512] = xb[N x 2080] @ W1b[2080 x 512]
  gemm_bf16<<<dim3(512 / 128, mTiles), T, 0, stream>>>(xb, KPAD, W1b, 512, h1, 512, N, KPAD);

  // layer-1 aggregation
  fill_f32<<<blks((long long)N * 512, T), T, 0, stream>>>(agg1, 0.0f, (long long)N * 512);
  scatter_edges<<<blks(El * 64, T), T, 0, stream>>>(h1, 512, agg1, 512, src_l, dst_l, El, 256, dinvl);
  scatter_edges<<<blks(Eg * 64, T), T, 0, stream>>>(h1 + 256, 512, agg1 + 256, 512, src_g, dst_g, Eg, 256, dinvg);
  finalize1<<<blks((long long)N * 512, T), T, 0, stream>>>(h1, agg1, dinvl, dinvg, b_l1, b_g1, h1b, N);

  // layer-2 GEMMs: [N x 256] @ [256 x 128], A strided into fused h1b
  gemm_bf16<<<dim3(1, mTiles), T, 0, stream>>>(h1b,       512, W2lb, 128, h2l, 128, N, 256);
  gemm_bf16<<<dim3(1, mTiles), T, 0, stream>>>(h1b + 256, 512, W2gb, 128, h2g, 128, N, 256);

  // layer-2 aggregation + pooling
  fill_f32<<<blks((long long)N * 128, T), T, 0, stream>>>(agg2l, 0.0f, (long long)N * 128);
  fill_f32<<<blks((long long)N * 128, T), T, 0, stream>>>(agg2g, 0.0f, (long long)N * 128);
  fill_f32<<<blks(NGRAPH * 256, T), T, 0, stream>>>(pooled, 0.0f, NGRAPH * 256);
  fill_f32<<<blks(NGRAPH, T), T, 0, stream>>>(cnt, 0.0f, NGRAPH);
  scatter_edges<<<blks(El * 32, T), T, 0, stream>>>(h2l, 128, agg2l, 128, src_l, dst_l, El, 128, dinvl);
  scatter_edges<<<blks(Eg * 32, T), T, 0, stream>>>(h2g, 128, agg2g, 128, src_g, dst_g, Eg, 128, dinvg);
  count_nodes<<<blks(N, T), T, 0, stream>>>(batch, cnt, N);
  finalize2_pool<<<blks((long long)N * 128, T), T, 0, stream>>>(h2l, agg2l, dinvl, b_l2, batch, pooled, 0, N);
  finalize2_pool<<<blks((long long)N * 128, T), T, 0, stream>>>(h2g, agg2g, dinvg, b_g2, batch, pooled, 128, N);

  // final fuse: [64 x 256] @ [256 x 128] + bias, ReLU
  fuse_out<<<NGRAPH, 128, 0, stream>>>(pooled, cnt, W_fuse, b_fuse, out);
}
